// Decoder_52123723104321
// MI455X (gfx1250) — compile-verified, matrix operand order checked by reading, not run on hardware
//
#include <hip/hip_runtime.h>
#include <hip/hip_bf16.h>

typedef __attribute__((ext_vector_type(16))) _Float16 v16h;
typedef __attribute__((ext_vector_type(8)))  float    v8f;

#define BATCH 4
#define CCH   256     // channels
#define HH    128
#define WW    128
#define HIN   32
#define WIN   32
#define NCLS  21
#define EPSV  1e-5f

// ---------------------------------------------------------------------------
// Kernel 1: pack conv weights  w[co][ci][kh][kw] (f32, OIHW 3x3)
//   -> wp[((tap*8 + cb)*256 + co)*32 + cilo]  (f16), tap = kh*3+kw, ci = cb*32+cilo
// ---------------------------------------------------------------------------
__global__ void pack_weights_k(const float* __restrict__ w, _Float16* __restrict__ wp) {
    int idx = blockIdx.x * blockDim.x + threadIdx.x;
    const int total = CCH * CCH * 9;
    if (idx >= total) return;
    int kw = idx % 3;
    int kh = (idx / 3) % 3;
    int ci = (idx / 9) % CCH;
    int co = idx / (9 * CCH);
    int tap = kh * 3 + kw;
    int cb  = ci >> 5;
    int cilo = ci & 31;
    wp[(((tap * 8 + cb) * CCH) + co) * 32 + cilo] = (_Float16)w[idx];
}

// ---------------------------------------------------------------------------
// Kernel 2: bilinear upsample (align_corners) 32->128 fused with dynamic 5x5
// smooth conv. Output NHWC f16.  One block per output pixel, 256 thr = chans.
// ---------------------------------------------------------------------------
__global__ void upsample_smooth_k(const float* __restrict__ x,
                                  const float* __restrict__ filt,
                                  _Float16* __restrict__ out) {
    const int pix = blockIdx.x;            // n*H*W + h*W + w
    const int w = pix & (WW - 1);
    const int h = (pix >> 7) & (HH - 1);
    const int n = pix >> 14;
    const int c = threadIdx.x;
    const float S = (float)(HIN - 1) / (float)(HH - 1);   // 31/127

    const float* xc = x + ((n * CCH + c) * HIN) * WIN;
    const float* kf = filt + ((((long)n * HH + h) * WW + w) * 25);

    float acc = 0.0f;
    #pragma unroll
    for (int i = 0; i < 5; ++i) {
        int yy = h + i - 2;
        if (yy < 0 || yy >= HH) continue;
        float fy = yy * S;
        int y0 = (int)fy; int y1 = min(y0 + 1, HIN - 1);
        float wy = fy - (float)y0;
        #pragma unroll
        for (int j = 0; j < 5; ++j) {
            int xx = w + j - 2;
            if (xx < 0 || xx >= WW) continue;
            float fx = xx * S;
            int x0 = (int)fx; int x1 = min(x0 + 1, WIN - 1);
            float wx = fx - (float)x0;
            float tl = xc[y0 * WIN + x0], tr = xc[y0 * WIN + x1];
            float bl = xc[y1 * WIN + x0], br = xc[y1 * WIN + x1];
            float t = tl + (tr - tl) * wx;
            float b = bl + (br - bl) * wx;
            acc = fmaf(kf[i * 5 + j], t + (b - t) * wy, acc);
        }
    }
    out[(long)pix * CCH + c] = (_Float16)acc;
}

// ---------------------------------------------------------------------------
// Async staging of one A-tile (3 rows x 66 pixels x 32 ci, f16) into LDS via
// GLOBAL_LOAD_ASYNC_TO_LDS_B128 (ASYNCcnt-tracked DMA, bypasses VGPRs).
// Each staging thread owns one 64B slot -> 4 async b128 ops (offsets 0..48
// apply to both LDS and global sides). OOB halo slots zero-filled with plain
// LDS stores. Per-wave ASYNCcnt += 4 per stage.
// ---------------------------------------------------------------------------
__device__ __forceinline__ void stage_tile_async(const _Float16* __restrict__ in,
                                                 int n, int h, int w0, int cb,
                                                 _Float16 (*buf)[66][32], int tid) {
    if (tid < 198) {
        const int r  = tid / 66;
        const int p  = tid % 66;
        const int hh = h - 1 + r;
        const int ww = w0 - 1 + p;
        _Float16* dst = &buf[r][p][0];
        if (hh >= 0 && hh < HH && ww >= 0 && ww < WW) {
            const _Float16* src = in + ((((long)n * HH + hh) * WW + ww) * CCH + cb * 32);
            // LDS aperture: low 32 bits of a generic LDS pointer are the
            // workgroup-relative LDS byte address (ISA 10.2).
            unsigned lo = (unsigned)(unsigned long long)(void*)dst;
            unsigned long long ga = (unsigned long long)(const void*)src;
            asm volatile("global_load_async_to_lds_b128 %0, %1, off"
                         :: "v"(lo), "v"(ga) : "memory");
            asm volatile("global_load_async_to_lds_b128 %0, %1, off offset:16"
                         :: "v"(lo), "v"(ga) : "memory");
            asm volatile("global_load_async_to_lds_b128 %0, %1, off offset:32"
                         :: "v"(lo), "v"(ga) : "memory");
            asm volatile("global_load_async_to_lds_b128 %0, %1, off offset:48"
                         :: "v"(lo), "v"(ga) : "memory");
        } else {
            uint4 z = {0u, 0u, 0u, 0u};
            uint4* d4 = (uint4*)dst;
            d4[0] = z; d4[1] = z; d4[2] = z; d4[3] = z;
        }
    }
}

// ---------------------------------------------------------------------------
// Kernel 3: 3x3 conv as implicit GEMM on WMMA f16->f32, fused BN + ReLU.
// Input/output NHWC f16.  Block = 256 thr (8 waves). Block tile: 64 pixels
// x 256 co. Wave tile: 64 px x 32 co = 4x2 accumulators. K loop: 8 ci-chunks
// of 32 x 9 taps. A tiles double-buffered in LDS via async DMA so the next
// chunk's staging overlaps the current chunk's 72 WMMAs.
// ---------------------------------------------------------------------------
__global__ void __launch_bounds__(256)
conv3x3_wmma_k(const _Float16* __restrict__ in,   // NHWC f16
               const _Float16* __restrict__ wp,   // packed weights
               const float* __restrict__ gamma, const float* __restrict__ beta,
               const float* __restrict__ mean,  const float* __restrict__ var,
               _Float16* __restrict__ out) {     // NHWC f16
    const int bw = blockIdx.x & 1;                 // which half-row
    const int h  = (blockIdx.x >> 1) & (HH - 1);
    const int n  = blockIdx.x >> 8;
    const int w0 = bw * 64;

    const int tid   = threadIdx.x;
    const int wave  = tid >> 5;
    const int lane  = tid & 31;
    const int lanelo = lane & 15;
    const int khalf  = (lane >> 4) << 4;           // 0 or 16 (contiguous K half, same for A and B)
    const int co_base = wave * 32;

    __shared__ __align__(64) _Float16 As[2][3][66][32];   // double-buffered tile

    v8f acc[4][2];
    #pragma unroll
    for (int m = 0; m < 4; ++m)
        #pragma unroll
        for (int j = 0; j < 2; ++j)
            acc[m][j] = {};

    stage_tile_async(in, n, h, w0, 0, As[0], tid);

    for (int cb = 0; cb < 8; ++cb) {
        const int cur = cb & 1;
        if (cb < 7) {
            stage_tile_async(in, n, h, w0, cb + 1, As[cur ^ 1], tid);
            // previous stage's 4 ops complete (in-order); next 4 may be in flight
            asm volatile("s_wait_asynccnt 0x4" ::: "memory");
        } else {
            asm volatile("s_wait_asynccnt 0x0" ::: "memory");
        }
        __syncthreads();

        #pragma unroll
        for (int dh = 0; dh < 3; ++dh) {
            #pragma unroll
            for (int dw = 0; dw < 3; ++dw) {
                const int tap = dh * 3 + dw;
                v16h bf[2];
                #pragma unroll
                for (int j = 0; j < 2; ++j) {
                    const int co = co_base + j * 16 + lanelo;
                    bf[j] = *(const v16h*)(wp +
                        ((((long)tap * 8 + cb) * CCH + co) * 32 + khalf));
                }
                #pragma unroll
                for (int m = 0; m < 4; ++m) {
                    v16h af = *(const v16h*)(&As[cur][dh][m * 16 + lanelo + dw][khalf]);
                    #pragma unroll
                    for (int j = 0; j < 2; ++j) {
                        acc[m][j] = __builtin_amdgcn_wmma_f32_16x16x32_f16(
                            false, af, false, bf[j], (short)0, acc[m][j],
                            false, false);
                    }
                }
            }
        }
        __syncthreads();   // all waves done reading before buffer reuse
    }

    // Epilogue: BN + ReLU, store NHWC f16.
    // C/D layout: lanes 0-15 -> M = r, lanes 16-31 -> M = r + 8; N = lane%16.
    const int mofs = (lane >> 4) << 3;  // 0 or 8
    #pragma unroll
    for (int j = 0; j < 2; ++j) {
        const int co = co_base + j * 16 + lanelo;
        const float sc = gamma[co] * rsqrtf(var[co] + EPSV);
        const float sh = beta[co] - mean[co] * sc;
        #pragma unroll
        for (int m = 0; m < 4; ++m) {
            #pragma unroll
            for (int r = 0; r < 8; ++r) {
                const int px = w0 + m * 16 + r + mofs;
                float v = fmaf(acc[m][j][r], sc, sh);
                v = fmaxf(v, 0.0f);
                out[(((long)n * HH + h) * WW + px) * CCH + co] = (_Float16)v;
            }
        }
    }
}

// ---------------------------------------------------------------------------
// Kernel 4: dynamic 5x5 smooth conv + 1x1 conv (256 -> 21) + bias.
// Input NHWC f16, output NCHW f32 (the final result).
// ---------------------------------------------------------------------------
__global__ void smooth_classify_k(const _Float16* __restrict__ in,  // NHWC f16
                                  const float* __restrict__ filt,
                                  const float* __restrict__ wlast,  // [21][256]
                                  const float* __restrict__ blast,  // [21]
                                  float* __restrict__ out) {        // NCHW f32
    const int pix = blockIdx.x;
    const int w = pix & (WW - 1);
    const int h = (pix >> 7) & (HH - 1);
    const int n = pix >> 14;
    const int c = threadIdx.x;

    const float* kf = filt + ((((long)n * HH + h) * WW + w) * 25);

    float acc = 0.0f;
    #pragma unroll
    for (int i = 0; i < 5; ++i) {
        int yy = h + i - 2;
        if (yy < 0 || yy >= HH) continue;
        #pragma unroll
        for (int j = 0; j < 5; ++j) {
            int xx = w + j - 2;
            if (xx < 0 || xx >= WW) continue;
            float v = (float)in[(((long)n * HH + yy) * WW + xx) * CCH + c];
            acc = fmaf(kf[i * 5 + j], v, acc);
        }
    }

    __shared__ float sred[CCH];
    sred[c] = acc;
    __syncthreads();

    // wave 0 performs the 21 dot products of length 256
    if (threadIdx.x < 32) {
        const int lane = threadIdx.x;
        float sv[8];
        #pragma unroll
        for (int q = 0; q < 8; ++q) sv[q] = sred[lane * 8 + q];
        for (int o = 0; o < NCLS; ++o) {
            float p = 0.0f;
            #pragma unroll
            for (int q = 0; q < 8; ++q)
                p = fmaf(sv[q], wlast[o * CCH + lane * 8 + q], p);
            #pragma unroll
            for (int off = 16; off > 0; off >>= 1)
                p += __shfl_down(p, off, 32);
            if (lane == 0)
                out[(((long)n * NCLS + o) * HH + h) * WW + w] = p + blast[o];
        }
    }
}

// ---------------------------------------------------------------------------
extern "C" void kernel_launch(void* const* d_in, const int* in_sizes, int n_in,
                              void* d_out, int out_size, void* d_ws, size_t ws_size,
                              hipStream_t stream) {
    const float* x       = (const float*)d_in[0];
    // d_in[1] (low_level_feat) only contributes its shape in the reference
    const float* filter4 = (const float*)d_in[2];
    const float* w_a     = (const float*)d_in[3];
    const float* gamma_a = (const float*)d_in[4];
    const float* beta_a  = (const float*)d_in[5];
    const float* mean_a  = (const float*)d_in[6];
    const float* var_a   = (const float*)d_in[7];
    const float* w_b     = (const float*)d_in[8];
    const float* gamma_b = (const float*)d_in[9];
    const float* beta_b  = (const float*)d_in[10];
    const float* mean_b  = (const float*)d_in[11];
    const float* var_b   = (const float*)d_in[12];
    const float* w_last  = (const float*)d_in[13];
    const float* b_last  = (const float*)d_in[14];
    float* out = (float*)d_out;

    char* ws = (char*)d_ws;
    const size_t WP_BYTES = (size_t)9 * 8 * CCH * 32 * sizeof(_Float16);       // 1.125 MB
    const size_t ACT_BYTES = (size_t)BATCH * HH * WW * CCH * sizeof(_Float16); // 32 MB
    _Float16* wpA = (_Float16*)(ws);
    _Float16* wpB = (_Float16*)(ws + WP_BYTES);
    _Float16* y0  = (_Float16*)(ws + 2 * WP_BYTES);
    _Float16* y1  = (_Float16*)(ws + 2 * WP_BYTES + ACT_BYTES);
    _Float16* y2  = (_Float16*)(ws + 2 * WP_BYTES + 2 * ACT_BYTES);

    const int wtotal = CCH * CCH * 9;
    pack_weights_k<<<(wtotal + 255) / 256, 256, 0, stream>>>(w_a, wpA);
    pack_weights_k<<<(wtotal + 255) / 256, 256, 0, stream>>>(w_b, wpB);

    const int npix = BATCH * HH * WW;                    // 65536 blocks
    upsample_smooth_k<<<npix, 256, 0, stream>>>(x, filter4, y0);

    const int nconv = BATCH * HH * 2;                    // 1024 blocks
    conv3x3_wmma_k<<<nconv, 256, 0, stream>>>(y0, wpA, gamma_a, beta_a, mean_a, var_a, y1);
    conv3x3_wmma_k<<<nconv, 256, 0, stream>>>(y1, wpB, gamma_b, beta_b, mean_b, var_b, y2);

    smooth_classify_k<<<npix, 256, 0, stream>>>(y2, filter4, w_last, b_last, out);
}
